// ConsciousnessStream_27187142983906
// MI455X (gfx1250) — compile-verified, matrix operand order checked by reading, not run on hardware
//
#include <hip/hip_runtime.h>
#include <stdint.h>

// ---------------------------------------------------------------------------
// ConsciousnessStream (gated GRU scan) for MI455X / gfx1250, wave32 + WMMA.
//
//  1) convert x, w_ih, w_hh, gate_w1 to bf16
//  2) gi  = x @ w_ih.T + b_ih       : WMMA bf16 GEMM (TDM-staged A tiles)
//  3) G1  = gelu(x @ gate_w1.T+b1)  : same GEMM template
//  4) g   = sigmoid(G1 @ w2.T + b2) : per-row wave reduction
//  5) scan: persistent 16-WG kernel, per step gh = h @ w_hh.T via WMMA
//     (w_hh bf16 = 6 MB -> L2 resident), fused GRU update, grid barrier.
//  6) stream = trace[:, T-S:, :] copy.
//
// GEMM A-tiles (128x32 bf16 = 8 KB) are loaded by the Tensor Data Mover
// (tensor_load_to_lds, TENSORcnt) into double-buffered LDS; fragments are
// then read with ds_load_b128. B operands stream from L2 via global b128.
// ---------------------------------------------------------------------------

#define B_ 32
#define T_ 2048
#define E_ 1024
#define S_ 24
#define N3_ 3072
#define NG_ 256
#define K_ 1024
#define MROWS (B_ * T_)   // 65536
#define NBLK_SCAN 16      // persistent blocks: 16*8 waves = 2 Mtiles * 64 Etiles

typedef __attribute__((ext_vector_type(16))) __bf16 v16bf;
typedef __attribute__((ext_vector_type(8)))  float  v8f;
typedef __attribute__((ext_vector_type(4)))  unsigned int v4u;
typedef __attribute__((ext_vector_type(8)))  int v8i_t;
typedef __attribute__((ext_vector_type(4)))  int v4i_t;

// ---- bf16 helpers ---------------------------------------------------------
__device__ __forceinline__ uint16_t f2bf(float f) {
  uint32_t x = __float_as_uint(f);
  uint32_t r = (x + 0x7FFFu + ((x >> 16) & 1u)) >> 16;   // round-nearest-even
  return (uint16_t)r;
}

// ---- WMMA fragment loaders ------------------------------------------------
// 16-bit A/B fragment (16x32 / 32x16), ISA 7.12.2 layout: lane&15 selects
// row(A)/col(B); lanes>=16 start at K=8; two b128 loads cover the 16 elems.
__device__ __forceinline__ v16bf load_frag_bf16(const uint16_t* base, int ld, int lane) {
  const uint16_t* p = base + (size_t)(lane & 15) * (size_t)ld + (size_t)((lane >> 4) << 3);
  uint4 lo = *(const uint4*)(p);
  uint4 hi = *(const uint4*)(p + 16);
  union { uint4 q[2]; v16bf v; } u;
  u.q[0] = lo; u.q[1] = hi;
  return u.v;
}

__device__ __forceinline__ v8f wmma_bf16(v16bf a, v16bf b, v8f c) {
  return __builtin_amdgcn_wmma_f32_16x16x32_bf16(false, a, false, b, (short)0, c,
                                                 false, false);
}

__device__ __forceinline__ float sigm(float x) { return 1.0f / (1.0f + __expf(-x)); }

// ---- f32 -> bf16 convert (grid stride) ------------------------------------
__global__ void cvt_bf16_kernel(const float* __restrict__ src,
                                uint16_t* __restrict__ dst, int n) {
  for (int i = blockIdx.x * blockDim.x + threadIdx.x; i < n;
       i += gridDim.x * blockDim.x)
    dst[i] = f2bf(src[i]);
}

// ---- init h0 from prev_stream[:, -1, :], zero barrier ---------------------
__global__ void init_kernel(const float* __restrict__ prev,
                            float* __restrict__ hf0, uint16_t* __restrict__ hb0,
                            int* __restrict__ bar) {
  int i = blockIdx.x * blockDim.x + threadIdx.x;   // 32768 threads
  if (i < B_ * E_) {
    int b = i >> 10, e = i & 1023;
    float v = prev[((size_t)b * S_ + (S_ - 1)) * E_ + e];
    hf0[i] = v;
    hb0[i] = f2bf(v);
  }
  if (i == 0) *bar = 0;
}

// ---- TDM: issue one 128x32 bf16 A-tile load into LDS ----------------------
// D# per CDNA5 ISA (2-D tile, groups 2/3 disabled):
//  g0 = { count=1 | flags=0, lds_addr, global_addr[31:0],
//         global_addr[56:32] | type(2)<<30 }
//  g1 = { wg_mask=0 | data_size(2B)<<16, tensor_dim0(K)[15:0]<<16,
//         0, tensor_dim1(M)[31:16] | tile_dim0(32)<<16,
//         tile_dim1(128), dim0_stride(K), 0, 0 }
// Toolchain here uses the 6-arg builtin (extra int32x8 group, zero for 2-D).
__device__ __forceinline__ void tdm_load_tileA(const uint16_t* gsrc,
                                               const uint16_t* lds_dst) {
  uint64_t ga = (uint64_t)(uintptr_t)gsrc;
  uint32_t lo = (uint32_t)(uintptr_t)lds_dst;     // low 32 bits = LDS offset
  v4u g0;
  g0.x = 1u;                                      // count=1, user descriptor
  g0.y = lo;
  g0.z = (uint32_t)ga;
  g0.w = (uint32_t)(ga >> 32) | 0x80000000u;      // type=2 ("image")
  v8i_t g1;
  g1[0] = 0x00010000;                             // data_size = 1 (2 bytes)
  g1[1] = (int)((uint32_t)(K_ & 0xFFFF) << 16);   // tensor_dim0 = K
  g1[2] = 0;                                      // dim0 hi / dim1 lo = 0
  g1[3] = (int)((MROWS >> 16) | (32u << 16));     // tensor_dim1 hi | tile_dim0
  g1[4] = 128;                                    // tile_dim1 = 128 rows
  g1[5] = K_;                                     // tensor_dim0_stride
  g1[6] = 0;
  g1[7] = 0;
  v4i_t z4 = {0, 0, 0, 0};
  v8i_t z8 = {0, 0, 0, 0, 0, 0, 0, 0};
  __builtin_amdgcn_tensor_load_to_lds(g0, g1, z4, z4, z8, 0);
}

// ---- big WMMA GEMM: C[M,N] = A[M,K] @ Bm[N,K]^T + bias (opt gelu) ---------
// Block tile 128x128, 8 waves as 2(M) x 4(N), wave tile 64x32.
// A tiles staged by TDM into double-buffered LDS; B direct from global/L2.
template <bool GELU>
__global__ __launch_bounds__(256)
void wmma_gemm_bias(const uint16_t* __restrict__ A, const uint16_t* __restrict__ Bm,
                    const float* __restrict__ bias, float* __restrict__ C,
                    int N) {
  __shared__ __align__(16) uint16_t shA[2][128 * 32];

  const int lane = threadIdx.x & 31;
  const int wave = threadIdx.x >> 5;
  const int wm = wave & 1;
  const int wn = wave >> 1;
  const int mblk = blockIdx.x * 128;
  const int n0 = blockIdx.y * 128 + wn * 32;

  const uint16_t* atile = A + (size_t)mblk * K_;
  const uint16_t* brow0 = Bm + (size_t)n0 * K_;
  const uint16_t* brow1 = Bm + (size_t)(n0 + 16) * K_;

  v8f acc[4][2] = {};

  if (wave == 0) tdm_load_tileA(atile, &shA[0][0]);

  for (int kt = 0; kt < K_ / 32; ++kt) {
    const int buf = kt & 1;
    if (wave == 0) __builtin_amdgcn_s_wait_tensorcnt(0);
    __syncthreads();                                   // tile[buf] ready; prev
    if (wave == 0 && kt + 1 < K_ / 32)                 // compute done -> safe
      tdm_load_tileA(atile + (kt + 1) * 32, &shA[buf ^ 1][0]);

    const int k = kt * 32;
    __builtin_prefetch(brow0 + k + 128, 0, 0);         // global_prefetch_b8
    v16bf bf0 = load_frag_bf16(brow0 + k, K_, lane);
    v16bf bf1 = load_frag_bf16(brow1 + k, K_, lane);
#pragma unroll
    for (int i = 0; i < 4; ++i) {
      // LDS fragment: row stride 32 elems, ds_load_b128 x2
      const uint16_t* p = &shA[buf][(wm * 64 + i * 16 + (lane & 15)) * 32 +
                                    ((lane >> 4) << 3)];
      union { uint4 q[2]; v16bf v; } u;
      u.q[0] = *(const uint4*)(p);
      u.q[1] = *(const uint4*)(p + 16);
      acc[i][0] = wmma_bf16(u.v, bf0, acc[i][0]);
      acc[i][1] = wmma_bf16(u.v, bf1, acc[i][1]);
    }
  }

  const int col = lane & 15;
  const int rbase = (lane >> 4) << 3;
#pragma unroll
  for (int i = 0; i < 4; ++i) {
#pragma unroll
    for (int j = 0; j < 2; ++j) {
      int n = n0 + j * 16 + col;
      float bi = bias[n];
#pragma unroll
      for (int r = 0; r < 8; ++r) {
        int m = mblk + wm * 64 + i * 16 + rbase + r;
        float v = acc[i][j][r] + bi;
        if (GELU) v = 0.5f * v * (1.0f + erff(v * 0.70710678118f));
        C[(size_t)m * N + n] = v;
      }
    }
  }
}

// ---- gate reduce: g[m] = sigmoid(sum_n G1[m,n]*w2[n] + b2) ----------------
__global__ __launch_bounds__(256)
void gate_reduce_kernel(const float* __restrict__ G1, const float* __restrict__ w2,
                        const float* __restrict__ b2, float* __restrict__ g) {
  int lane = threadIdx.x & 31;
  int row = blockIdx.x * 8 + (threadIdx.x >> 5);     // one wave per row
  const float* p = G1 + (size_t)row * NG_;
  float s = 0.f;
#pragma unroll
  for (int j = 0; j < NG_ / 32; ++j) s += p[lane + j * 32] * w2[lane + j * 32];
#pragma unroll
  for (int off = 16; off > 0; off >>= 1) s += __shfl_xor(s, off, 32);
  if (lane == 0) g[row] = sigm(s + b2[0]);
}

// ---- grid barrier (monotonic counter, agent-scope fences) -----------------
__device__ __forceinline__ void grid_sync(int* bar, int step) {
  __syncthreads();
  if (threadIdx.x == 0) {
    __builtin_amdgcn_fence(__ATOMIC_RELEASE, "agent");   // global_wb
    __hip_atomic_fetch_add(bar, 1, __ATOMIC_RELAXED, __HIP_MEMORY_SCOPE_AGENT);
    const int target = NBLK_SCAN * (step + 1);
    while (__hip_atomic_load(bar, __ATOMIC_RELAXED, __HIP_MEMORY_SCOPE_AGENT) < target)
      __builtin_amdgcn_s_sleep(2);
    __builtin_amdgcn_fence(__ATOMIC_ACQUIRE, "agent");   // global_inv
  }
  __syncthreads();
}

// ---- persistent scan: per step gh = h @ w_hh^T (WMMA) + fused GRU update --
// 128 waves total: wave w -> mtile = w&1 (batch rows), etile = w>>1 (E cols).
__global__ __launch_bounds__(256)
void scan_kernel(const float* __restrict__ gi,      // (B*T, 3E), includes b_ih
                 const float* __restrict__ gvals,   // (B*T)
                 const uint16_t* __restrict__ whh,  // (3E, E) bf16
                 const float* __restrict__ bhh,     // (3E)
                 float* __restrict__ hf0, float* __restrict__ hf1,
                 uint16_t* __restrict__ hb0, uint16_t* __restrict__ hb1,
                 int* __restrict__ bar,
                 float* __restrict__ trace) {       // (B, T, E)
  const int lane = threadIdx.x & 31;
  const int w = (blockIdx.x << 3) + (threadIdx.x >> 5);   // 0..127
  const int mtile = w & 1;
  const int etile = w >> 1;
  const int ebase = etile * 16;

  const uint16_t* wr = whh + (size_t)(ebase)*E_;
  const uint16_t* wz = whh + (size_t)(ebase + E_) * E_;
  const uint16_t* wn = whh + (size_t)(ebase + 2 * E_) * E_;

  const int col = ebase + (lane & 15);
  const float bhr = bhh[col], bhz = bhh[col + E_], bhn = bhh[col + 2 * E_];
  const int rbase = ((lane >> 4) << 3) + mtile * 16;

  for (int t = 0; t < T_; ++t) {
    const uint16_t* hb = (t & 1) ? hb1 : hb0;
    const float* hf = (t & 1) ? hf1 : hf0;
    uint16_t* hbn = (t & 1) ? hb0 : hb1;
    float* hfn = (t & 1) ? hf0 : hf1;

    v8f ar = {}, az = {}, an = {};
    const uint16_t* arow = hb + (size_t)(mtile * 16) * E_;
#pragma unroll 4
    for (int k = 0; k < K_; k += 32) {
      v16bf af = load_frag_bf16(arow + k, E_, lane);
      v16bf br = load_frag_bf16(wr + k, E_, lane);
      v16bf bz = load_frag_bf16(wz + k, E_, lane);
      v16bf bn = load_frag_bf16(wn + k, E_, lane);
      ar = wmma_bf16(af, br, ar);
      az = wmma_bf16(af, bz, az);
      an = wmma_bf16(af, bn, an);
    }

#pragma unroll
    for (int r = 0; r < 8; ++r) {
      int b = rbase + r;                       // batch row 0..31
      size_t rowg = (size_t)b * T_ + t;
      const float* gir = gi + rowg * N3_;
      float vr = sigm(gir[col] + ar[r] + bhr);
      float vz = sigm(gir[col + E_] + az[r] + bhz);
      float vn = tanhf(gir[col + 2 * E_] + vr * (an[r] + bhn));
      float hold = hf[b * E_ + col];
      float gate = gvals[rowg];
      float nh = (1.0f - vz) * vn + vz * hold;
      float up = gate * nh + (1.0f - gate) * hold;
      trace[rowg * E_ + col] = up;
      hfn[b * E_ + col] = up;
      hbn[b * E_ + col] = f2bf(up);
    }
    grid_sync(bar, t);
  }
}

// ---- stream = trace[:, T-S:, :] -------------------------------------------
__global__ void stream_copy_kernel(const float* __restrict__ trace,
                                   float* __restrict__ stream_out) {
  int i = blockIdx.x * blockDim.x + threadIdx.x;   // B*S*E threads
  if (i >= B_ * S_ * E_) return;
  int e = i & 1023;
  int s = (i >> 10) % S_;
  int b = i / (S_ * E_);
  stream_out[i] = trace[((size_t)b * T_ + (T_ - S_ + s)) * E_ + e];
}

// ---------------------------------------------------------------------------
extern "C" void kernel_launch(void* const* d_in, const int* in_sizes, int n_in,
                              void* d_out, int out_size, void* d_ws, size_t ws_size,
                              hipStream_t stream) {
  const float* x    = (const float*)d_in[0];
  const float* prev = (const float*)d_in[1];
  const float* gw1  = (const float*)d_in[2];
  const float* gb1  = (const float*)d_in[3];
  const float* gw2  = (const float*)d_in[4];
  const float* gb2  = (const float*)d_in[5];
  const float* wih  = (const float*)d_in[6];
  const float* whh  = (const float*)d_in[7];
  const float* bih  = (const float*)d_in[8];
  const float* bhh  = (const float*)d_in[9];

  float* stream_out = (float*)d_out;
  float* trace = stream_out + (size_t)B_ * S_ * E_;

  // workspace carve (256B aligned)
  char* p = (char*)d_ws;
  auto carve = [&](size_t bytes) {
    char* r = p;
    p += (bytes + 255) & ~(size_t)255;
    return r;
  };
  uint16_t* xb   = (uint16_t*)carve((size_t)MROWS * E_ * 2);
  uint16_t* wihb = (uint16_t*)carve((size_t)N3_ * E_ * 2);
  uint16_t* whhb = (uint16_t*)carve((size_t)N3_ * E_ * 2);
  uint16_t* gw1b = (uint16_t*)carve((size_t)NG_ * E_ * 2);
  float*    gi   = (float*)carve((size_t)MROWS * N3_ * 4);
  float*    G1   = (float*)carve((size_t)MROWS * NG_ * 4);
  float*    gv   = (float*)carve((size_t)MROWS * 4);
  float*    hf0  = (float*)carve((size_t)B_ * E_ * 4);
  float*    hf1  = (float*)carve((size_t)B_ * E_ * 4);
  uint16_t* hb0  = (uint16_t*)carve((size_t)B_ * E_ * 2);
  uint16_t* hb1  = (uint16_t*)carve((size_t)B_ * E_ * 2);
  int*      bar  = (int*)carve(256);

  // 1) bf16 conversions
  cvt_bf16_kernel<<<4096, 256, 0, stream>>>(x, xb, MROWS * E_);
  cvt_bf16_kernel<<<1024, 256, 0, stream>>>(wih, wihb, N3_ * E_);
  cvt_bf16_kernel<<<1024, 256, 0, stream>>>(whh, whhb, N3_ * E_);
  cvt_bf16_kernel<<<256, 256, 0, stream>>>(gw1, gw1b, NG_ * E_);

  // 2) h0 + barrier init
  init_kernel<<<(B_ * E_ + 255) / 256, 256, 0, stream>>>(prev, hf0, hb0, bar);

  // 3) gi = x @ w_ih^T + b_ih   (65536 x 3072 x 1024)
  wmma_gemm_bias<false><<<dim3(MROWS / 128, N3_ / 128), 256, 0, stream>>>(
      xb, wihb, bih, gi, N3_);

  // 4) G1 = gelu(x @ gate_w1^T + gate_b1)   (65536 x 256 x 1024)
  wmma_gemm_bias<true><<<dim3(MROWS / 128, NG_ / 128), 256, 0, stream>>>(
      xb, gw1b, gb1, G1, NG_);

  // 5) g = sigmoid(G1 @ w2^T + b2)
  gate_reduce_kernel<<<MROWS / 8, 256, 0, stream>>>(G1, gw2, gb2, gv);

  // 6) sequential scan, persistent WMMA kernel
  scan_kernel<<<NBLK_SCAN, 256, 0, stream>>>(gi, gv, whhb, bhh, hf0, hf1, hb0,
                                             hb1, bar, trace);

  // 7) stream slice
  stream_copy_kernel<<<(B_ * S_ * E_ + 255) / 256, 256, 0, stream>>>(trace,
                                                                     stream_out);
}
